// MultiHeadAttention_54958401519898
// MI455X (gfx1250) — matrix operand order repaired and checked by reading, not compiled, not run on hardware
//
#include <hip/hip_runtime.h>
#include <hip/hip_bf16.h>

// ---- CDNA5 WMMA types ------------------------------------------------------
typedef __attribute__((ext_vector_type(16))) __bf16 v16bf;
typedef __attribute__((ext_vector_type(8)))  float  v8f;

union Frag {
    v16bf v;
    uint4 q[2];
};

__device__ __forceinline__ unsigned short f2bf_bits(float f) {
    unsigned u = __builtin_bit_cast(unsigned, f);
    unsigned r = u + 0x7FFFu + ((u >> 16) & 1u);  // round-to-nearest-even
    return (unsigned short)(r >> 16);
}

// ---- async global->LDS copy path (gfx1250) ---------------------------------
#if defined(__HIP_DEVICE_COMPILE__) && defined(__gfx1250__) && \
    __has_builtin(__builtin_amdgcn_global_load_async_to_lds_b128)
#define ASYNC_CP 1
#else
#define ASYNC_CP 0
#endif

#if ASYNC_CP
typedef int v4i_cp __attribute__((__vector_size__(4 * sizeof(int))));
typedef __attribute__((address_space(1))) v4i_cp* gbl_v4p;
typedef __attribute__((address_space(3))) v4i_cp* lds_v4p;
#endif

// Copy 16 bytes (8 bf16) global -> LDS.
__device__ __forceinline__ void cp16(const unsigned short* g, unsigned short* l) {
#if ASYNC_CP
    __builtin_amdgcn_global_load_async_to_lds_b128(
        (gbl_v4p)(unsigned long long)g,
        (lds_v4p)(unsigned)(unsigned long long)l,
        0, 0);
#else
    *(uint4*)l = *(const uint4*)g;
#endif
}

__device__ __forceinline__ void wait_stage() {
#if ASYNC_CP
#if __has_builtin(__builtin_amdgcn_s_wait_asynccnt)
    __builtin_amdgcn_s_wait_asynccnt(0);
#else
    asm volatile("s_wait_asynccnt 0x0" ::: "memory");
#endif
#endif
}

// ---- Generic all-bf16 WMMA GEMM --------------------------------------------
// C[z] = A[z] * B[z]  (+ bias | *scale + mask), out f32 or bf16 (optionally
// stored transposed). Block tile 256(M) x 64(N) x 32(K); 256 threads = 8
// waves; each wave owns a 32x64 strip = 2x4 accumulators of 16x16, fed by
// v_wmma_f32_16x16x32_bf16. Double-buffered async LDS staging.
template <bool OUT_BF16, bool MASKSCALE, bool BIAS, bool STORE_T>
__global__ __launch_bounds__(256)
void gemm_wmma(const unsigned short* __restrict__ A, int lda, long long aBatch,
               const unsigned short* __restrict__ B, int ldb, long long bBatch,
               const float* __restrict__ bias,
               const float* __restrict__ mask, long long maskBatch,
               void* __restrict__ Cp, int ldc, long long cBatch,
               int M, int N, int K, float scale) {
    constexpr int BM = 256, BN = 64, BK = 32;
    constexpr int LDA_S = BK + 8;   // 40 shorts -> 80B row (16B-aligned chunks)
    constexpr int LDB_S = BN + 8;   // 72 shorts -> 144B row
    __shared__ __align__(16) unsigned short As[2][BM * LDA_S];
    __shared__ __align__(16) unsigned short Bs[2][BK * LDB_S];

    const int tid  = threadIdx.x;
    const int wave = tid >> 5;
    const int lane = tid & 31;
    const int z    = blockIdx.z;
    const int m0   = blockIdx.y * BM;
    const int n0   = blockIdx.x * BN;

    const unsigned short* Ab = A + (size_t)z * aBatch + (size_t)m0 * lda;
    const unsigned short* Bb = B + (size_t)z * bBatch + n0;

    // stage one 256x32 A tile + 32x64 B tile at K offset k0 into buffer buf
    auto stage = [&](int k0, int buf) {
        // A: 1024 x 16B chunks, 4 per thread
#pragma unroll
        for (int it = 0; it < 4; ++it) {
            const int id = tid + it * 256;
            const int r = id >> 2, c = id & 3;  // row 0..255, chunk 0..3
            cp16(Ab + (size_t)r * lda + k0 + c * 8,
                 &As[buf][r * LDA_S + c * 8]);
        }
        // B: 256 x 16B chunks, 1 per thread
        {
            const int r = tid >> 3, c = tid & 7;  // k-row 0..31, chunk 0..7
            cp16(Bb + (size_t)(k0 + r) * ldb + c * 8,
                 &Bs[buf][r * LDB_S + c * 8]);
        }
    };

    v8f acc[2][4];
#pragma unroll
    for (int h = 0; h < 2; ++h)
#pragma unroll
        for (int j = 0; j < 4; ++j) acc[h][j] = (v8f)0.0f;

    stage(0, 0);
    wait_stage();
    __syncthreads();

    int buf = 0;
    for (int k0 = 0; k0 < K; k0 += BK) {
        if (k0 + BK < K) stage(k0 + BK, buf ^ 1);

        // ---- fragments per documented CDNA5 16-bit layouts -----------------
        // A 16x32: lane = M row; VGPR0-3 hold K=koff..+7, VGPR4-7 hold K=+16
        const unsigned short* Al = As[buf];
        const unsigned short* Bl = Bs[buf];
        const int mr   = wave * 32 + (lane & 15);
        const int koff = (lane >> 4) * 8;
        Frag a0, a1;
        a0.q[0] = *(const uint4*)&Al[mr * LDA_S + koff];
        a0.q[1] = *(const uint4*)&Al[mr * LDA_S + koff + 16];
        a1.q[0] = *(const uint4*)&Al[(mr + 16) * LDA_S + koff];
        a1.q[1] = *(const uint4*)&Al[(mr + 16) * LDA_S + koff + 16];
#pragma unroll
        for (int j = 0; j < 4; ++j) {
            // B 32x16: lane = N col; VGPR0-3 hold K=koff..+7, VGPR4-7 hold K=+16
            Frag b;
#pragma unroll
            for (int s = 0; s < 16; ++s)
                b.v[s] = __builtin_bit_cast(__bf16, Bl[(koff + (s & 7) + (s >> 3) * 16) * LDB_S + j * 16 + (lane & 15)]);
            acc[0][j] = __builtin_amdgcn_wmma_f32_16x16x32_bf16(
                false, a0.v, false, b.v, (short)0, acc[0][j], false, false);
            acc[1][j] = __builtin_amdgcn_wmma_f32_16x16x32_bf16(
                false, a1.v, false, b.v, (short)0, acc[1][j], false, false);
        }
        wait_stage();
        __syncthreads();
        buf ^= 1;
    }

    // ---- epilogue: C/D layout -> VGPR r: M = r + (lane>=16)*8, N = lane&15
    const int nl = lane & 15;
    const int mb = (lane >> 4) * 8;
#pragma unroll
    for (int h = 0; h < 2; ++h) {
#pragma unroll
        for (int j = 0; j < 4; ++j) {
            const int ng = n0 + j * 16 + nl;
            float bv = 0.0f;
            if constexpr (BIAS) bv = bias[ng];
#pragma unroll
            for (int r = 0; r < 8; ++r) {
                const int mg = m0 + wave * 32 + h * 16 + mb + r;
                float c = acc[h][j][r];
                if constexpr (MASKSCALE)
                    c = c * scale +
                        mask[(size_t)z * maskBatch + (size_t)mg * ldc + ng];
                else if constexpr (BIAS)
                    c += bv;
                if constexpr (STORE_T)   // store C^T (bf16): C[n, m]
                    ((unsigned short*)Cp)[(size_t)z * cBatch +
                                          (size_t)ng * ldc + mg] = f2bf_bits(c);
                else if constexpr (OUT_BF16)
                    ((unsigned short*)Cp)[(size_t)z * cBatch +
                                          (size_t)mg * ldc + ng] = f2bf_bits(c);
                else
                    ((float*)Cp)[(size_t)z * cBatch + (size_t)mg * ldc + ng] = c;
            }
        }
    }
}

// ---- f32 -> bf16 bulk convert (8 elems / thread) ---------------------------
__global__ __launch_bounds__(256)
void cvt_f32_bf16(const float* __restrict__ src, unsigned short* __restrict__ dst,
                  long long n) {
    long long i = ((long long)blockIdx.x * 256 + threadIdx.x) * 8;
    if (i + 8 > n) return;
    float4 a = *(const float4*)(src + i);
    float4 b = *(const float4*)(src + i + 4);
    unsigned short o[8] = {f2bf_bits(a.x), f2bf_bits(a.y), f2bf_bits(a.z),
                           f2bf_bits(a.w), f2bf_bits(b.x), f2bf_bits(b.y),
                           f2bf_bits(b.z), f2bf_bits(b.w)};
    *(uint4*)(dst + i) = *(const uint4*)o;
}

// ---- Row softmax: f32 scores row -> bf16 P written in place ----------------
// One 256-thread block per row of 2048; P row stride = 4096 bf16 elements.
__global__ __launch_bounds__(256)
void softmax_bf16(float* __restrict__ scores, int S) {
    const int tid = threadIdx.x;
    float* rp = scores + (size_t)blockIdx.x * S;
    __shared__ float red[256];

    float v[8];
    float mx = -3.4e38f;
#pragma unroll
    for (int i = 0; i < 8; ++i) {
        v[i] = rp[tid + 256 * i];
        mx = fmaxf(mx, v[i]);
    }
    red[tid] = mx;
    __syncthreads();
    for (int s = 128; s > 0; s >>= 1) {
        if (tid < s) red[tid] = fmaxf(red[tid], red[tid + s]);
        __syncthreads();
    }
    mx = red[0];
    __syncthreads();

    float sum = 0.0f;
#pragma unroll
    for (int i = 0; i < 8; ++i) {
        v[i] = __expf(v[i] - mx);
        sum += v[i];
    }
    red[tid] = sum;
    __syncthreads();
    for (int s = 128; s > 0; s >>= 1) {
        if (tid < s) red[tid] += red[tid + s];
        __syncthreads();
    }
    const float inv = 1.0f / red[0];
    __syncthreads();  // all f32 reads of this row are done; safe to overwrite

    unsigned short* pp = (unsigned short*)rp;  // bf16 row, stride 4096 elems
#pragma unroll
    for (int i = 0; i < 8; ++i)
        pp[tid + 256 * i] = f2bf_bits(v[i] * inv);
}

// ---- Host launcher ---------------------------------------------------------
extern "C" void kernel_launch(void* const* d_in, const int* in_sizes, int n_in,
                              void* d_out, int out_size, void* d_ws, size_t ws_size,
                              hipStream_t stream) {
    constexpr int Bn = 4, S = 2048, H = 1024;
    const float* hs   = (const float*)d_in[0];
    const float* mask = (const float*)d_in[1];
    const float* Wq   = (const float*)d_in[2];
    const float* bq   = (const float*)d_in[3];
    const float* Wk   = (const float*)d_in[4];
    const float* bk   = (const float*)d_in[5];
    const float* Wv   = (const float*)d_in[6];
    const float* bv   = (const float*)d_in[7];

    // ws layout (bf16 unless noted):
    //   hsb[8192x1024] | wq|wk|wv[1024x1024] | Q[8192x1024] | K^T[4][1024x2048]
    //   | V[8192x1024] | scores f32 [4][2048x2048] (reused in place as bf16 P)
    unsigned short* hsb = (unsigned short*)d_ws;
    unsigned short* wqb = hsb + (size_t)Bn * S * H;
    unsigned short* wkb = wqb + (size_t)H * H;
    unsigned short* wvb = wkb + (size_t)H * H;
    unsigned short* qb  = wvb + (size_t)H * H;
    unsigned short* ktb = qb + (size_t)Bn * S * H;
    unsigned short* vb  = ktb + (size_t)Bn * S * H;
    float* scores       = (float*)(vb + (size_t)Bn * S * H);

    const dim3 blk(256);
    const long long SH = (long long)S * H;
    const long long SS = (long long)S * S;

    // 0) one-time f32->bf16 conversions (hs read once instead of 3x)
    cvt_f32_bf16<<<dim3((Bn * S * H) / 2048), blk, 0, stream>>>(hs, hsb,
                                                                (long long)Bn * SH);
    cvt_f32_bf16<<<dim3((H * H) / 2048), blk, 0, stream>>>(Wq, wqb, (long long)H * H);
    cvt_f32_bf16<<<dim3((H * H) / 2048), blk, 0, stream>>>(Wk, wkb, (long long)H * H);
    cvt_f32_bf16<<<dim3((H * H) / 2048), blk, 0, stream>>>(Wv, wvb, (long long)H * H);

    // 1) projections: Q,V normal; K stored transposed per batch -> K^T[d][s]
    const dim3 gp(H / 64, (Bn * S) / 256, 1);
    gemm_wmma<true, false, true, false><<<gp, blk, 0, stream>>>(
        hsb, H, 0, wqb, H, 0, bq, nullptr, 0, qb, H, 0, Bn * S, H, H, 1.0f);
    gemm_wmma<true, false, true, false><<<gp, blk, 0, stream>>>(
        hsb, H, 0, wvb, H, 0, bv, nullptr, 0, vb, H, 0, Bn * S, H, H, 1.0f);
    const dim3 gk(H / 64, S / 256, Bn);
    gemm_wmma<true, false, true, true><<<gk, blk, 0, stream>>>(
        hsb, H, SH, wkb, H, 0, bk, nullptr, 0, ktb, S, SH, S, H, H, 1.0f);

    // 2) scores = Q * K^T * (1/sqrt(H)) + mask   (per batch, B row-major)
    const dim3 gs(S / 64, S / 256, Bn);
    gemm_wmma<false, true, false, false><<<gs, blk, 0, stream>>>(
        qb, H, SH, ktb, S, SH, nullptr, mask, SS, scores, S, SS, S, S, H,
        0.03125f);

    // 3) softmax rows -> bf16 P in place (row stride 4096 bf16)
    softmax_bf16<<<dim3(Bn * S), blk, 0, stream>>>(scores, S);

    // 4) out = P * V  (per batch) -> f32 d_out
    const unsigned short* P = (const unsigned short*)scores;
    const dim3 go(H / 64, S / 256, Bn);
    gemm_wmma<false, false, false, false><<<go, blk, 0, stream>>>(
        P, 2 * S, (long long)S * 2 * S, vb, H, SH, nullptr, nullptr, 0,
        (float*)d_out, H, SH, S, H, S, 1.0f);
}
